// LM_47588237640184
// MI455X (gfx1250) — compile-verified, matrix operand order checked by reading, not run on hardware
//
#include <hip/hip_runtime.h>

// ---------------- problem sizes ----------------
#define V_   32000
#define E_   1024
#define H_   1024
#define B_   64
#define G4H  (4*H_)
#define TX_  128   // T1-1 steps

typedef __attribute__((ext_vector_type(16))) __bf16 v16bf;
typedef __attribute__((ext_vector_type(8)))  __bf16 v8bf;
typedef __attribute__((ext_vector_type(8)))  float  v8f;

// Load one 16x32 (A) or 32x16 (B) bf16 WMMA fragment for this lane from a
// row-major [rows x ldk] matrix. rp = pointer to this lane's row.
// 16-bit fragment striping: lanes 0-15 hold K = kc+[0..7] and kc+16+[0..7],
// lanes 16-31 hold K = kc+8+[0..7] and kc+24+[0..7]  -> sel = lane>>4.
__device__ __forceinline__ v16bf load_frag(const __bf16* rp, int kc, int sel) {
    v8bf lo = *(const v8bf*)(rp + kc + sel * 8);
    v8bf hi = *(const v8bf*)(rp + kc + 16 + sel * 8);
    return __builtin_shufflevector(lo, hi, 0,1,2,3,4,5,6,7,8,9,10,11,12,13,14,15);
}

__device__ __forceinline__ v8f wmma_bf16(v16bf a, v16bf b, v8f c) {
    return __builtin_amdgcn_wmma_f32_16x16x32_bf16(false, a, false, b, (short)0, c, false, false);
}

// Accumulate NT adjacent 16x16 n-tiles sharing one A fragment per K chunk.
// pa: this lane's A row; pb: this lane's B row for n-tile 0 (row-major N x K, ld=ldk).
template<int NT>
__device__ __forceinline__ void gemm_strip(const __bf16* pa, const __bf16* pb, int ldk,
                                           int K, int sel, v8f acc[NT]) {
    for (int kc = 0; kc < K; kc += 32) {
        v16bf a = load_frag(pa, kc, sel);
        #pragma unroll
        for (int j = 0; j < NT; ++j) {
            v16bf b = load_frag(pb + (size_t)j * 16 * ldk, kc, sel);
            acc[j] = wmma_bf16(a, b, acc[j]);
        }
    }
}

// ---------------- prep kernels ----------------
__global__ void k_cast_bf16(const float* __restrict__ s, __bf16* __restrict__ d, size_t n) {
    size_t i = (size_t)blockIdx.x * blockDim.x + threadIdx.x;
    if (i < n) d[i] = (__bf16)s[i];
}

// src: [R x C] f32 row-major  ->  dst: [C x R] bf16 row-major
__global__ void k_transpose_cast(const float* __restrict__ src, __bf16* __restrict__ dst,
                                 int R, int C) {
    __shared__ float tile[32][33];
    int c0 = blockIdx.x * 32, r0 = blockIdx.y * 32;
    int tx = threadIdx.x, ty = threadIdx.y;       // block (32,8)
    #pragma unroll
    for (int i = 0; i < 32; i += 8)
        tile[ty + i][tx] = src[(size_t)(r0 + ty + i) * C + (c0 + tx)];
    __syncthreads();
    #pragma unroll
    for (int i = 0; i < 32; i += 8)
        dst[(size_t)(c0 + ty + i) * R + (r0 + tx)] = (__bf16)tile[tx][ty + i];
}

__global__ void k_init(float* h, float* c, __bf16* hbf, float* out) {
    int i = blockIdx.x * blockDim.x + threadIdx.x;   // covers B_*H_
    h[i] = 0.f; c[i] = 0.f; hbf[i] = (__bf16)0.0f;
    if (i == 0) out[0] = 0.f;
}

// ---------------- gates GEMM: g = emb[x] @ W_ih + h @ W_hh ----------------
// NT=2; block 128 (4 waves, 32 cols each); grid (G4H/128, B_/16)
__global__ void k_gates(const int* __restrict__ tok, const __bf16* __restrict__ embb,
                        const __bf16* __restrict__ WihT, const __bf16* __restrict__ WhhT,
                        const __bf16* __restrict__ hbf, float* __restrict__ g) {
    int wave = threadIdx.x >> 5, lane = threadIdx.x & 31;
    int sel = lane >> 4, lr = lane & 15;
    int n0 = (blockIdx.x * 4 + wave) * 32;
    int mt = blockIdx.y;
    const __bf16* pax = embb + (size_t)tok[mt * 16 + lr] * E_;
    const __bf16* pah = hbf  + (size_t)(mt * 16 + lr) * H_;
    const __bf16* pbx = WihT + (size_t)(n0 + lr) * E_;
    const __bf16* pbh = WhhT + (size_t)(n0 + lr) * H_;
    v8f acc[2] = {};
    gemm_strip<2>(pax, pbx, E_, E_, sel, acc);
    gemm_strip<2>(pah, pbh, H_, H_, sel, acc);
    #pragma unroll
    for (int j = 0; j < 2; ++j) {
        int col = n0 + j * 16 + lr;
        #pragma unroll
        for (int r = 0; r < 8; ++r)
            g[(size_t)(mt * 16 + sel * 8 + r) * G4H + col] = acc[j][r];
    }
}

// ---------------- LSTM pointwise update ----------------
__global__ void k_lstm(const float* __restrict__ g, const float* __restrict__ bias,
                       float* __restrict__ c, float* __restrict__ h,
                       __bf16* __restrict__ hbf, const float* __restrict__ mrow) {
    int idx = blockIdx.x * blockDim.x + threadIdx.x;   // B_*H_
    int b = idx >> 10, hh = idx & (H_ - 1);
    size_t base = (size_t)b * G4H + hh;
    float gi = g[base]            + bias[hh];
    float gf = g[base + H_]       + bias[H_ + hh];
    float gc = g[base + 2 * H_]   + bias[2 * H_ + hh];
    float go = g[base + 3 * H_]   + bias[3 * H_ + hh];
    float i_ = 1.f / (1.f + __expf(-gi));
    float f_ = 1.f / (1.f + __expf(-gf));
    float o_ = 1.f / (1.f + __expf(-go));
    float gg = tanhf(gc);
    float cn = f_ * c[idx] + i_ * gg;
    float hn = o_ * tanhf(cn);
    float m  = mrow[b];
    float h2 = m * hn + (1.f - m) * h[idx];
    float c2 = m * cn + (1.f - m) * c[idx];
    h[idx] = h2; c[idx] = c2; hbf[idx] = (__bf16)h2;
}

// ---------------- projection GEMM: out = h @ Wr^T + br  (bf16 out) ----------------
// NT=2; block 128; grid (E_/128, B_/16)
__global__ void k_proj(const __bf16* __restrict__ hbf, const __bf16* __restrict__ Wrb,
                       const float* __restrict__ br, __bf16* __restrict__ outb) {
    int wave = threadIdx.x >> 5, lane = threadIdx.x & 31;
    int sel = lane >> 4, lr = lane & 15;
    int n0 = (blockIdx.x * 4 + wave) * 32;
    int mt = blockIdx.y;
    const __bf16* pa = hbf + (size_t)(mt * 16 + lr) * H_;
    const __bf16* pb = Wrb + (size_t)(n0 + lr) * H_;
    v8f acc[2] = {};
    gemm_strip<2>(pa, pb, H_, H_, sel, acc);
    #pragma unroll
    for (int j = 0; j < 2; ++j) {
        int col = n0 + j * 16 + lr;
        float bb = br[col];
        #pragma unroll
        for (int r = 0; r < 8; ++r)
            outb[(size_t)(mt * 16 + sel * 8 + r) * E_ + col] = (__bf16)(acc[j][r] + bb);
    }
}

// ---------------- logits GEMM: logits = out @ emb^T + bd ----------------
// NT=4; block 128 (4 waves, 64 cols each); grid (V_/256, B_/16)
__global__ void k_logits(const __bf16* __restrict__ outb, const __bf16* __restrict__ embb,
                         const float* __restrict__ bd, float* __restrict__ logits) {
    int wave = threadIdx.x >> 5, lane = threadIdx.x & 31;
    int sel = lane >> 4, lr = lane & 15;
    int n0 = (blockIdx.x * 4 + wave) * 64;
    int mt = blockIdx.y;
    const __bf16* pa = outb + (size_t)(mt * 16 + lr) * E_;
    const __bf16* pb = embb + (size_t)(n0 + lr) * E_;
    v8f acc[4] = {};
    gemm_strip<4>(pa, pb, E_, E_, sel, acc);
    #pragma unroll
    for (int j = 0; j < 4; ++j) {
        int col = n0 + j * 16 + lr;
        float bb = bd[col];
        #pragma unroll
        for (int r = 0; r < 8; ++r)
            logits[(size_t)(mt * 16 + sel * 8 + r) * V_ + col] = acc[j][r] + bb;
    }
}

// ---------------- per-row log-softmax NLL, accumulated into out[0] ----------------
__global__ void k_loss(const float* __restrict__ logits, const int* __restrict__ ytok,
                       const float* __restrict__ mrow, float* __restrict__ out) {
    __shared__ float red[256];
    int b = blockIdx.x, t = threadIdx.x;
    const float* row = logits + (size_t)b * V_;
    float mx = -3.4e38f;
    for (int v = t; v < V_; v += 256) mx = fmaxf(mx, row[v]);
    red[t] = mx; __syncthreads();
    for (int s = 128; s > 0; s >>= 1) { if (t < s) red[t] = fmaxf(red[t], red[t + s]); __syncthreads(); }
    mx = red[0]; __syncthreads();
    float sum = 0.f;
    for (int v = t; v < V_; v += 256) sum += __expf(row[v] - mx);
    red[t] = sum; __syncthreads();
    for (int s = 128; s > 0; s >>= 1) { if (t < s) red[t] += red[t + s]; __syncthreads(); }
    if (t == 0) {
        float nll = mx + __logf(red[0]) - row[ytok[b]];
        atomicAdd(out, nll * mrow[b] * (1.0f / ((float)B_ * (float)B_)));
    }
}

// ---------------- host driver ----------------
extern "C" void kernel_launch(void* const* d_in, const int* in_sizes, int n_in,
                              void* d_out, int out_size, void* d_ws, size_t ws_size,
                              hipStream_t stream) {
    (void)in_sizes; (void)n_in; (void)out_size; (void)ws_size;
    const int*   data = (const int*)  d_in[0];  // (129, 64)
    const float* mask = (const float*)d_in[1];  // (129, 64)
    const float* emb  = (const float*)d_in[2];  // (V, E)
    const float* Wih  = (const float*)d_in[3];  // (E, 4H)
    const float* Whh  = (const float*)d_in[4];  // (H, 4H)
    const float* bvec = (const float*)d_in[5];  // (4H,)
    const float* Wr   = (const float*)d_in[6];  // (E, H)
    const float* br   = (const float*)d_in[7];  // (E,)
    const float* bd   = (const float*)d_in[8];  // (V,)
    float* out = (float*)d_out;

    char* p = (char*)d_ws;
    auto alloc = [&](size_t bytes) { char* r = p; p += (bytes + 255) & ~(size_t)255; return r; };
    __bf16* embb   = (__bf16*)alloc((size_t)V_ * E_ * 2);     // 65.5 MB, L2-resident
    __bf16* WihT   = (__bf16*)alloc((size_t)G4H * E_ * 2);    // 8.4 MB (transposed)
    __bf16* WhhT   = (__bf16*)alloc((size_t)G4H * H_ * 2);    // 8.4 MB (transposed)
    __bf16* Wrb    = (__bf16*)alloc((size_t)E_ * H_ * 2);     // 2 MB (already N x K)
    __bf16* hbf    = (__bf16*)alloc((size_t)B_ * H_ * 2);
    __bf16* outb   = (__bf16*)alloc((size_t)B_ * E_ * 2);
    float*  g      = (float*) alloc((size_t)B_ * G4H * 4);
    float*  hm     = (float*) alloc((size_t)B_ * H_ * 4);
    float*  cm     = (float*) alloc((size_t)B_ * H_ * 4);
    float*  logits = (float*) alloc((size_t)B_ * V_ * 4);

    // one-time prep (runs every call; deterministic)
    {
        size_t n = (size_t)V_ * E_;
        k_cast_bf16<<<dim3((unsigned)((n + 255) / 256)), 256, 0, stream>>>(emb, embb, n);
    }
    k_transpose_cast<<<dim3(G4H / 32, E_ / 32), dim3(32, 8), 0, stream>>>(Wih, WihT, E_, G4H);
    k_transpose_cast<<<dim3(G4H / 32, H_ / 32), dim3(32, 8), 0, stream>>>(Whh, WhhT, H_, G4H);
    {
        size_t n = (size_t)E_ * H_;
        k_cast_bf16<<<dim3((unsigned)((n + 255) / 256)), 256, 0, stream>>>(Wr, Wrb, n);
    }
    k_init<<<dim3(B_ * H_ / 256), 256, 0, stream>>>(hm, cm, hbf, out);

    for (int t = 0; t < TX_; ++t) {
        const int*   xt = data + (size_t)t * B_;
        const int*   yt = data + (size_t)(t + 1) * B_;
        const float* mt = mask + (size_t)(t + 1) * B_;
        k_gates <<<dim3(G4H / 128, B_ / 16), 128, 0, stream>>>(xt, embb, WihT, WhhT, hbf, g);
        k_lstm  <<<dim3(B_ * H_ / 256), 256, 0, stream>>>(g, bvec, cm, hm, hbf, mt);
        k_proj  <<<dim3(E_ / 128, B_ / 16), 128, 0, stream>>>(hbf, Wrb, br, outb);
        k_logits<<<dim3(V_ / 256, B_ / 16), 128, 0, stream>>>(outb, embb, bd, logits);
        k_loss  <<<dim3(B_), 256, 0, stream>>>(logits, yt, mt, out);
    }
}
